// RaftSLAM_47064251630152
// MI455X (gfx1250) — compile-verified, matrix operand order checked by reading, not run on hardware
//
#include <hip/hip_runtime.h>
#include <cstdint>
#include <cstddef>

// ---------------------------------------------------------------------------
// RAFT-SLAM GRU update block for MI455X (gfx1250, wave32, WMMA).
// Convs = implicit GEMM via v_wmma_f32_16x16x32_f16:
//   - A-fragments: cooperative LDS staging in hardware fragment order
//   - B-fragments: per-wave async global->LDS copies (global_load_async_to_lds_b128
//     + s_wait_asynccnt), double buffered, software pipelined
// ---------------------------------------------------------------------------

typedef __attribute__((ext_vector_type(16))) _Float16 v16h;
typedef __attribute__((ext_vector_type(8)))  float    v8f;
typedef int gv4i __attribute__((vector_size(16)));   // matches builtin pointee

#define AS1 __attribute__((address_space(1)))
#define AS3 __attribute__((address_space(3)))

#define BN_ 24
#define H_  60
#define W_  80
#define P_  (H_ * W_)   // 4800 pixels per image (multiple of 64; 16 | W)

#if __has_builtin(__builtin_amdgcn_global_load_async_to_lds_b128) && \
    __has_builtin(__builtin_amdgcn_s_wait_asynccnt)
#define USE_ASYNC_B 1
#else
#define USE_ASYNC_B 0
#endif

// CDNA5 16-bit A/B matrix 16x32 fragment layout (ISA 7.12.2):
// element (m, k): lane = m + 16*((k>>3)&1), half j = (k>>4)*8 + (k&7)
__device__ __host__ __forceinline__ int frag_k_of(int lane, int j) {
  int v = j >> 1;
  return ((v >> 2) << 4) + ((lane >> 4) << 3) + ((v & 3) << 1) + (j & 1);
}

// --------------------------- weight pre-pack -------------------------------
// Pack conv weight (Cout, Cin*KH*KW) f32 into B-fragment-ordered f16 blocks:
// block (kt, nt) holds 512 halfs so lane L owns halfs [L*16, L*16+16) as one
// contiguous 32B chunk (async-copy friendly). Out-of-range k/n -> 0.
__global__ void pack_w_kernel(const float* __restrict__ src, _Float16* __restrict__ dst,
                              int K, int Cout, int nTiles, int total) {
  int idx = blockIdx.x * blockDim.x + threadIdx.x;
  if (idx >= total) return;
  int within = idx & 511;
  int tile   = idx >> 9;
  int nt = tile % nTiles;
  int kt = tile / nTiles;
  int lane = within >> 4;
  int j    = within & 15;
  int n  = lane & 15;
  int k  = kt * 32 + frag_k_of(lane, j);
  int co = nt * 16 + n;
  float val = 0.0f;
  if (k < K && co < Cout) val = src[(size_t)co * K + k];
  dst[idx] = (_Float16)val;
}

// --------------------------- f32 -> f16 slice copy -------------------------
__global__ void cvt_slice_kernel(const float* __restrict__ src, _Float16* __restrict__ dst,
                                 int inner, size_t dstImgStride, size_t dstOff, int total) {
  int idx = blockIdx.x * blockDim.x + threadIdx.x;
  if (idx >= total) return;
  int img = idx / inner;
  int rem = idx - img * inner;
  dst[(size_t)img * dstImgStride + dstOff + (size_t)rem] = (_Float16)src[idx];
}

// --------------------------- implicit-GEMM conv ----------------------------
// Grid: (P/64, BN). Block: 256 threads = 8 waves. Wave w computes output
// channels [16w, 16w+16) for a 64-pixel tile (4 sub-tiles of 16 pixels).
// Per K-step of 32:
//   - 4 A-fragments staged cooperatively in LDS in fragment order (dbl-buffered)
//   - each wave's B-fragment streamed global->LDS by the async engine
//   - 4 WMMAs per wave per barrier
// act: 0=none 1=relu 2=sigmoid 3=tanh
template<int KH, int KW, int PAD>
__global__ __launch_bounds__(256) void conv_wmma_kernel(
    const _Float16* __restrict__ x,     // (BN, Cin, H, W) f16
    const _Float16* __restrict__ wpk,   // packed weights
    const float*    __restrict__ bias,  // (Cout) f32
    _Float16*       __restrict__ out16, // optional f16 out
    float*          __restrict__ out32, // optional f32 out
    int Cin, int Cout, size_t out16ImgStride, size_t out32ImgStride, int act) {

  const int HW  = P_;
  const int KHW = KH * KW;
  __shared__ __align__(32) _Float16 shA[2][4][512];   // 8 KB
  __shared__ __align__(32) _Float16 shB[8][2][512];   // 16 KB

  const int tid  = threadIdx.x;
  const int wave = tid >> 5;
  const int lane = tid & 31;
  const int mBase = blockIdx.x * 64;
  const int img   = blockIdx.y;
  const int nTiles = (Cout + 15) >> 4;
  const bool compute = (wave < nTiles);

  const int Ktot   = Cin * KHW;
  const int ksteps = (Ktot + 31) >> 5;
  const _Float16* xin = x + (size_t)img * Cin * HW;

  // Each 16-pixel sub-tile lies in a single image row (16 | W):
  int phs[4], pws[4];
#pragma unroll
  for (int mt = 0; mt < 4; ++mt) {
    int pb = mBase + mt * 16;
    phs[mt] = pb / W_;
    pws[mt] = pb - phs[mt] * W_;
  }

  // staging role: thread covers k = tid>>3 and pixels (tid&7)*2, (tid&7)*2+1
  const int sk   = tid >> 3;                       // 0..31 (k within K step)
  const int sm0  = (tid & 7) << 1;                 // 0,2,..,14
  const int sj   = ((sk >> 4) << 3) + (sk & 7);    // fragment half index
  const int sHiK = (sk >> 3) & 1;                  // lane-half from k

  auto stageA = [&](int ks, int buf) {
    int kk = (ks << 5) + sk;
    bool kvalid = (kk < Ktot);
    int ci = 0, r = 0, s = 0;
    if (kvalid) {
      ci = kk / KHW;                 // KHW compile-time: strength-reduced
      int rem = kk - ci * KHW;
      r = rem / KW;
      s = rem - r * KW;
    }
    const size_t chanBase = (size_t)ci * HW;
#pragma unroll
    for (int mt = 0; mt < 4; ++mt) {
      int ih = phs[mt] + r - PAD;
      bool rowOk = kvalid && ih >= 0 && ih < H_;
      const size_t rowBase = chanBase + (size_t)ih * W_;
#pragma unroll
      for (int t = 0; t < 2; ++t) {
        int mm = sm0 + t;
        int iw = pws[mt] + mm + s - PAD;
        _Float16 val = (_Float16)0.0f;
        if (rowOk && iw >= 0 && iw < W_) val = xin[rowBase + iw];
        shA[buf][mt][((mm + (sHiK << 4)) << 4) + sj] = val;
      }
    }
  };

#if USE_ASYNC_B
  auto asyncB = [&](int ks, int buf) {
    // lane-contiguous 32B: perfect 2x b128 async copy, tracked by ASYNCcnt
    const _Float16* src = wpk + ((size_t)(ks * nTiles + wave) << 9) + (lane << 4);
    const _Float16* dst = &shB[wave][buf][lane << 4];
    AS1 gv4i* gp = (AS1 gv4i*)(uintptr_t)src;
    AS3 gv4i* lp = (AS3 gv4i*)(uint32_t)(uintptr_t)dst;
    __builtin_amdgcn_global_load_async_to_lds_b128(gp,     lp,     0, 0);
    __builtin_amdgcn_global_load_async_to_lds_b128(gp + 1, lp + 1, 0, 0);
  };
#endif

  stageA(0, 0);
#if USE_ASYNC_B
  if (compute) asyncB(0, 0);
#endif
  __syncthreads();

  v8f acc[4] = {};

  for (int ks = 0; ks < ksteps; ++ks) {
    int buf = ks & 1;
    if (compute) {
      v16h b;
#if USE_ASYNC_B
      __builtin_amdgcn_s_wait_asynccnt(0);     // B(ks) landed in LDS
      asm volatile("" ::: "memory");           // keep ds read below the wait
      b = *(const v16h*)&shB[wave][buf][lane << 4];
      if (ks + 1 < ksteps) asyncB(ks + 1, buf ^ 1);   // pipeline next B
#else
      b = *(const v16h*)(wpk + ((size_t)(ks * nTiles + wave) << 9) + (lane << 4));
#endif
#pragma unroll
      for (int mt = 0; mt < 4; ++mt) {
        v16h a = *(const v16h*)&shA[buf][mt][lane << 4];
        acc[mt] = __builtin_amdgcn_wmma_f32_16x16x32_f16(false, a, false, b,
                                                         (short)0, acc[mt], false, false);
      }
    }
    if (ks + 1 < ksteps) stageA(ks + 1, buf ^ 1);
    __syncthreads();
  }

  // ---- epilogue: D layout n = lane&15, m = vgpr + 8*(lane>>4)
  if (compute) {
    int n  = lane & 15;
    int hi = lane >> 4;
    int co = (wave << 4) + n;
    if (co < Cout) {
      float bs = bias[co];
#pragma unroll
      for (int mt = 0; mt < 4; ++mt) {
#pragma unroll
        for (int v = 0; v < 8; ++v) {
          int mm = v + (hi << 3);
          float val = acc[mt][v] + bs;
          if      (act == 1) val = fmaxf(val, 0.0f);
          else if (act == 2) val = 1.0f / (1.0f + __expf(-val));
          else if (act == 3) val = tanhf(val);
          int px = mBase + mt * 16 + mm;
          if (out16) out16[(size_t)img * out16ImgStride + (size_t)co * HW + px] = (_Float16)val;
          if (out32) out32[(size_t)img * out32ImgStride + (size_t)co * HW + px] = val;
        }
      }
    }
  }
}

// --------------------------- GRU elementwise -------------------------------
__global__ void gru_rh_kernel(_Float16* __restrict__ xcat, const _Float16* __restrict__ r,
                              int total) {
  int idx = blockIdx.x * blockDim.x + threadIdx.x;
  if (idx >= total) return;
  const int inner = 128 * P_;
  int img = idx / inner;
  int rem = idx - img * inner;
  size_t xi = (size_t)img * (448 * P_) + rem;
  float h  = (float)xcat[xi];
  float rr = (float)r[idx];
  xcat[xi] = (_Float16)(h * rr);
}

__global__ void gru_combine_kernel(const float* __restrict__ hsrc,
                                   const _Float16* __restrict__ z,
                                   const _Float16* __restrict__ q,
                                   float* __restrict__ outNet,
                                   _Float16* __restrict__ h16, int total) {
  int idx = blockIdx.x * blockDim.x + threadIdx.x;
  if (idx >= total) return;
  float zz = (float)z[idx];
  float hn = (1.0f - zz) * hsrc[idx] + zz * (float)q[idx];
  outNet[idx] = hn;
  h16[idx]    = (_Float16)hn;
}

__global__ void head_tr_kernel(const float* __restrict__ src, float* __restrict__ dst,
                               int total) {
  int idx = blockIdx.x * blockDim.x + threadIdx.x;
  if (idx >= total) return;
  int img = idx / (3 * P_);
  int rem = idx - img * (3 * P_);
  int c   = rem / P_;
  int px  = rem - c * P_;
  dst[(size_t)img * (P_ * 3) + (size_t)px * 3 + c] = src[idx];
}

// ---------------------------------------------------------------------------
extern "C" void kernel_launch(void* const* d_in, const int* in_sizes, int n_in,
                              void* d_out, int out_size, void* d_ws, size_t ws_size,
                              hipStream_t stream) {
  (void)in_sizes; (void)n_in; (void)out_size; (void)ws_size;

  const float* net  = (const float*)d_in[0];
  const float* inp  = (const float*)d_in[1];
  const float* corr = (const float*)d_in[2];
  const float* flow = (const float*)d_in[3];
  const float* w_ce1 = (const float*)d_in[4];  const float* b_ce1 = (const float*)d_in[5];
  const float* w_ce2 = (const float*)d_in[6];  const float* b_ce2 = (const float*)d_in[7];
  const float* w_fe1 = (const float*)d_in[8];  const float* b_fe1 = (const float*)d_in[9];
  const float* w_fe2 = (const float*)d_in[10]; const float* b_fe2 = (const float*)d_in[11];
  const float* w_z   = (const float*)d_in[12]; const float* b_z   = (const float*)d_in[13];
  const float* w_r   = (const float*)d_in[14]; const float* b_r   = (const float*)d_in[15];
  const float* w_q   = (const float*)d_in[16]; const float* b_q   = (const float*)d_in[17];
  const float* w_d1  = (const float*)d_in[18]; const float* b_d1  = (const float*)d_in[19];
  const float* w_d2  = (const float*)d_in[20]; const float* b_d2  = (const float*)d_in[21];
  const float* w_w1  = (const float*)d_in[22]; const float* b_w1  = (const float*)d_in[23];
  const float* w_w2  = (const float*)d_in[24]; const float* b_w2  = (const float*)d_in[25];

  const int n128      = BN_ * 128 * P_;       // 14,745,600
  const int corrTotal = BN_ * 197 * P_;       // 22,694,400
  const int flTotal   = BN_ * 3 * P_;         //    345,600
  const int xcatTotal = BN_ * 448 * P_;       // 51,609,600

  char* base = (char*)d_ws;
  size_t off = 0;
  auto alloc16 = [&](size_t elems) -> _Float16* {
    _Float16* p = (_Float16*)(base + off); off += elems * sizeof(_Float16); return p;
  };
  auto alloc32 = [&](size_t elems) -> float* {
    float* p = (float*)(base + off); off += elems * sizeof(float); return p;
  };
  auto pkElems = [](int K, int Co) -> size_t {
    return (size_t)((K + 31) / 32) * ((Co + 15) / 16) * 512;
  };

  _Float16* xcat     = alloc16(xcatTotal);   // [h | inp | corr_feat | flow_feat]
  _Float16* corr16   = alloc16(corrTotal);
  _Float16* fl16     = alloc16(flTotal);
  _Float16* bufA     = alloc16(n128);        // c1 -> f1 -> hnew16
  _Float16* bufZ     = alloc16(n128);        // z  -> d1
  _Float16* bufR     = alloc16(n128);        // r  -> q -> w1
  float*    deltaRaw = alloc32(flTotal);
  float*    weightRaw= alloc32(flTotal);

  _Float16* pk_ce1 = alloc16(pkElems(197,  128));
  _Float16* pk_ce2 = alloc16(pkElems(1152, 128));
  _Float16* pk_fe1 = alloc16(pkElems(147,  128));
  _Float16* pk_fe2 = alloc16(pkElems(1152, 64));
  _Float16* pk_z   = alloc16(pkElems(4032, 128));
  _Float16* pk_r   = alloc16(pkElems(4032, 128));
  _Float16* pk_q   = alloc16(pkElems(4032, 128));
  _Float16* pk_d1  = alloc16(pkElems(1152, 128));
  _Float16* pk_d2  = alloc16(pkElems(1152, 3));
  _Float16* pk_w1  = alloc16(pkElems(1152, 128));
  _Float16* pk_w2  = alloc16(pkElems(1152, 3));

  auto pack = [&](const float* w, _Float16* dst, int K, int Co) {
    int nT = (Co + 15) / 16;
    int total = (int)pkElems(K, Co);
    pack_w_kernel<<<(total + 255) / 256, 256, 0, stream>>>(w, dst, K, Co, nT, total);
  };
  pack(w_ce1, pk_ce1, 197,  128);
  pack(w_ce2, pk_ce2, 1152, 128);
  pack(w_fe1, pk_fe1, 147,  128);
  pack(w_fe2, pk_fe2, 1152, 64);
  pack(w_z,   pk_z,   4032, 128);
  pack(w_r,   pk_r,   4032, 128);
  pack(w_q,   pk_q,   4032, 128);
  pack(w_d1,  pk_d1,  1152, 128);
  pack(w_d2,  pk_d2,  1152, 3);
  pack(w_w1,  pk_w1,  1152, 128);
  pack(w_w2,  pk_w2,  1152, 3);

  auto cvt = [&](const float* s, _Float16* d, int inner, size_t stride, size_t dOff, int total) {
    cvt_slice_kernel<<<(total + 255) / 256, 256, 0, stream>>>(s, d, inner, stride, dOff, total);
  };
  cvt(net,  xcat,   128 * P_, (size_t)448 * P_, 0,                 n128);
  cvt(inp,  xcat,   128 * P_, (size_t)448 * P_, (size_t)128 * P_,  n128);
  cvt(corr, corr16, 197 * P_, (size_t)197 * P_, 0,                 corrTotal);
  cvt(flow, fl16,   3 * P_,   (size_t)3 * P_,   0,                 flTotal);

  dim3 cg(P_ / 64, BN_);
  const size_t S128 = (size_t)128 * P_;
  const size_t S448 = (size_t)448 * P_;
  const size_t S3   = (size_t)3 * P_;

  // correlation encoder: 1x1 197->128 relu ; 3x3 128->128 relu (into xcat[256:384])
  conv_wmma_kernel<1,1,0><<<cg, 256, 0, stream>>>(corr16, pk_ce1, b_ce1,
      bufA, nullptr, 197, 128, S128, 0, 1);
  conv_wmma_kernel<3,3,1><<<cg, 256, 0, stream>>>(bufA, pk_ce2, b_ce2,
      xcat + (size_t)256 * P_, nullptr, 128, 128, S448, 0, 1);
  // flow encoder: 7x7 3->128 relu ; 3x3 128->64 relu (into xcat[384:448])
  conv_wmma_kernel<7,7,3><<<cg, 256, 0, stream>>>(fl16, pk_fe1, b_fe1,
      bufA, nullptr, 3, 128, S128, 0, 1);
  conv_wmma_kernel<3,3,1><<<cg, 256, 0, stream>>>(bufA, pk_fe2, b_fe2,
      xcat + (size_t)384 * P_, nullptr, 128, 64, S448, 0, 1);

  // ConvGRU gates (448 -> 128)
  conv_wmma_kernel<3,3,1><<<cg, 256, 0, stream>>>(xcat, pk_z, b_z,
      bufZ, nullptr, 448, 128, S128, 0, 2);                      // z = sigmoid
  conv_wmma_kernel<3,3,1><<<cg, 256, 0, stream>>>(xcat, pk_r, b_r,
      bufR, nullptr, 448, 128, S128, 0, 2);                      // r = sigmoid
  gru_rh_kernel<<<(n128 + 255) / 256, 256, 0, stream>>>(xcat, bufR, n128);
  conv_wmma_kernel<3,3,1><<<cg, 256, 0, stream>>>(xcat, pk_q, b_q,
      bufR, nullptr, 448, 128, S128, 0, 3);                      // q = tanh
  gru_combine_kernel<<<(n128 + 255) / 256, 256, 0, stream>>>(
      net, bufZ, bufR, (float*)d_out, bufA, n128);               // net_out + hnew16

  // delta head: 3x3 128->128 relu ; 3x3 128->3
  conv_wmma_kernel<3,3,1><<<cg, 256, 0, stream>>>(bufA, pk_d1, b_d1,
      bufZ, nullptr, 128, 128, S128, 0, 1);
  conv_wmma_kernel<3,3,1><<<cg, 256, 0, stream>>>(bufZ, pk_d2, b_d2,
      nullptr, deltaRaw, 128, 3, 0, S3, 0);
  // weight head: 3x3 128->128 relu ; 3x3 128->3 sigmoid
  conv_wmma_kernel<3,3,1><<<cg, 256, 0, stream>>>(bufA, pk_w1, b_w1,
      bufR, nullptr, 128, 128, S128, 0, 1);
  conv_wmma_kernel<3,3,1><<<cg, 256, 0, stream>>>(bufR, pk_w2, b_w2,
      nullptr, weightRaw, 128, 3, 0, S3, 2);

  float* outF = (float*)d_out;
  head_tr_kernel<<<(flTotal + 255) / 256, 256, 0, stream>>>(deltaRaw,  outF + n128,            flTotal);
  head_tr_kernel<<<(flTotal + 255) / 256, 256, 0, stream>>>(weightRaw, outF + n128 + flTotal,  flTotal);
}
